// YOLOv2Loss_44796508898037
// MI455X (gfx1250) — compile-verified, matrix operand order checked by reading, not run on hardware
//
#include <hip/hip_runtime.h>

#define WDIM   26
#define HDIM   26
#define HWDIM  (WDIM * HDIM)   // 676
#define ADIM   5
#define NCLS   20
#define CHS    (5 + NCLS)      // 25
#define IGNORE_T 0.5f

typedef float    v2f  __attribute__((ext_vector_type(2)));
typedef float    v8f  __attribute__((ext_vector_type(8)));
typedef _Float16 v16h __attribute__((ext_vector_type(16)));

#if defined(__gfx1250__) && __has_builtin(__builtin_amdgcn_global_load_async_to_lds_b32)
#define HAVE_ASYNC_LDS 1
#else
#define HAVE_ASYNC_LDS 0
#endif

__device__ __forceinline__ float sigf(float x) { return 1.0f / (1.0f + expf(-x)); }

__device__ __forceinline__ float iou_xywh(float ax, float ay, float aw, float ah,
                                          float bx, float by, float bw, float bh) {
  float tlx = fmaxf(ax - aw * 0.5f, bx - bw * 0.5f);
  float tly = fmaxf(ay - ah * 0.5f, by - bh * 0.5f);
  float brx = fminf(ax + aw * 0.5f, bx + bw * 0.5f);
  float bry = fminf(ay + ah * 0.5f, by + bh * 0.5f);
  float iw = fmaxf(brx - tlx, 0.0f);
  float ih = fmaxf(bry - tly, 0.0f);
  float inter = iw * ih;
  return inter / (aw * ah + bw * bh - inter);
}

// Fold the 16 distinct values (mirrored across lane halves) down to the full
// sum using ds_swizzle xor patterns (group-of-32 encoding: (xor<<5)|0x1F).
// Avoids the ds_bpermute + index-clamp sequence __shfl_xor lowers to.
__device__ __forceinline__ float fold16_sum(float r) {
#if defined(__AMDGCN__) && __has_builtin(__builtin_amdgcn_ds_swizzle)
  r += __int_as_float(__builtin_amdgcn_ds_swizzle(__float_as_int(r), 0x11F)); // xor 8
  r += __int_as_float(__builtin_amdgcn_ds_swizzle(__float_as_int(r), 0x09F)); // xor 4
  r += __int_as_float(__builtin_amdgcn_ds_swizzle(__float_as_int(r), 0x05F)); // xor 2
  r += __int_as_float(__builtin_amdgcn_ds_swizzle(__float_as_int(r), 0x03F)); // xor 1
#else
  r += __shfl_xor(r, 8, 32);
  r += __shfl_xor(r, 4, 32);
  r += __shfl_xor(r, 2, 32);
  r += __shfl_xor(r, 1, 32);
#endif
  return r;
}

// Wave32 sum-reduction. V_WMMA_F32_16X16X4_F32 with an all-ones A matrix:
// D[0][n] = column sums of B, so one WMMA folds the 32 lane partials (B VGPR0
// across lanes) into 16 column sums held in D VGPR0 lanes 0..15 (mirrored in
// lanes 16..31 as row M=8). fold16_sum finishes. Exact in f32; correct for any
// K/N lane partition of B since all 16 columns get summed.
__device__ __forceinline__ float wave_reduce_sum(float v) {
#if defined(__gfx1250__) && __has_builtin(__builtin_amdgcn_wmma_f32_16x16x4_f32)
  v2f a; a[0] = 1.0f; a[1] = 1.0f;        // A = ones(16x4)
  v2f b; b[0] = v;    b[1] = 0.0f;        // one partial per lane in B VGPR0
  v8f c = {};
  c = __builtin_amdgcn_wmma_f32_16x16x4_f32(false, a, false, b, (short)0, c, false, false);
  return fold16_sum(c[0]);
#elif defined(__gfx1250__) && __has_builtin(__builtin_amdgcn_wmma_f32_16x16x32_f16)
  v16h a, b;
#pragma unroll
  for (int i = 0; i < 16; ++i) { a[i] = (_Float16)1.0f; b[i] = (_Float16)0.0f; }
  b[0] = (_Float16)v;
  v8f c = {};
  c = __builtin_amdgcn_wmma_f32_16x16x32_f16(false, a, false, b, (short)0, c, false, false);
  return fold16_sum(c[0]);
#else
  for (int off = 16; off > 0; off >>= 1) v += __shfl_xor(v, off, 32);
  return v;
#endif
}

__global__ void yolo_init(int* __restrict__ winner, int* __restrict__ has_pos,
                          int ntot, int B) {
  int i = blockIdx.x * blockDim.x + threadIdx.x;
  if (i < ntot) winner[i] = -1;
  if (i < B) has_pos[i] = 0;
}

// One block per batch, one thread per gt object.
__global__ void yolo_prep(const float* __restrict__ targets, const float* __restrict__ anchors,
                          float* __restrict__ gt, int* __restrict__ gcls,
                          int* __restrict__ cellA, int* __restrict__ aidxA,
                          int* __restrict__ num_obj, int M) {
  int b = blockIdx.x;
  int m = threadIdx.x;
  __shared__ int cnt;
  if (m == 0) cnt = 0;
  __syncthreads();

  const float* t = targets + ((size_t)b * M + m) * 5;
  float c0 = t[0], x = t[1], y = t[2], w = t[3], h = t[4];
  if (c0 + x + y + w + h > 0.0f) atomicAdd(&cnt, 1);

  float gx = x * (float)WDIM, gy = y * (float)HDIM;
  float gw = w * (float)WDIM, gh = h * (float)HDIM;
  size_t gi = ((size_t)b * M + m) * 4;
  gt[gi + 0] = gx; gt[gi + 1] = gy; gt[gi + 2] = gw; gt[gi + 3] = gh;
  gcls[b * M + m] = (int)c0;

  int cell = (int)(floorf(gy) * (float)WDIM + floorf(gx));
  cell = min(max(cell, 0), HWDIM - 1);

  // argmax_a IoU(anchor box centered at cell center, gt); first-max wins.
  float acx = (float)(cell % WDIM) + 0.5f;
  float acy = (float)(cell / WDIM) + 0.5f;
  float best = -1.0f; int bi = 0;
#pragma unroll
  for (int a = 0; a < ADIM; ++a) {
    float v = iou_xywh(acx, acy, anchors[2 * a], anchors[2 * a + 1], gx, gy, gw, gh);
    if (v > best) { best = v; bi = a; }
  }
  cellA[b * M + m] = cell;
  aidxA[b * M + m] = bi;

  __syncthreads();
  if (m == 0) num_obj[b] = cnt;
}

// grid = (ceil(A*HW/256), B). idx -> a = idx/HW (coalesced along hw).
__global__ void yolo_maxiou(const float* __restrict__ outputs, const float* __restrict__ anchors,
                            const float* __restrict__ gt, float* __restrict__ max_iou,
                            int* __restrict__ has_pos, int M) {
  int b = blockIdx.y;
  __shared__ float gts[512];  // up to M=128 boxes * 4 floats

#if HAVE_ASYNC_LDS
  // CDNA5 async copy: global -> LDS without touching VGPRs (ASYNCcnt).
  for (int i = threadIdx.x; i < M * 4; i += blockDim.x) {
    const float* src = gt + (size_t)b * M * 4 + i;
    __builtin_amdgcn_global_load_async_to_lds_b32(
        (__attribute__((address_space(1))) int*)src,
        (__attribute__((address_space(3))) int*)&gts[i],
        0, 0);
  }
#if __has_builtin(__builtin_amdgcn_s_wait_asynccnt)
  __builtin_amdgcn_s_wait_asynccnt(0);
#else
  asm volatile("s_wait_asynccnt 0" ::: "memory");
#endif
#else
  for (int i = threadIdx.x; i < M * 4; i += blockDim.x)
    gts[i] = gt[(size_t)b * M * 4 + i];
#endif
  __syncthreads();

  int idx = blockIdx.x * blockDim.x + threadIdx.x;
  if (idx >= ADIM * HWDIM) return;
  int a  = idx / HWDIM;
  int hw = idx - a * HWDIM;
  int x = hw % WDIM, y = hw / WDIM;

  const float* base = outputs + ((size_t)b * (ADIM * CHS) + (size_t)a * CHS) * HWDIM + hw;
  float aw = anchors[2 * a], ah = anchors[2 * a + 1];
  float px = sigf(base[0]) + (float)x;
  float py = sigf(base[HWDIM]) + (float)y;
  float pw = expf(base[2 * HWDIM]) * aw;
  float ph = expf(base[3 * HWDIM]) * ah;

  // Invalid (all-zero) gt rows yield IoU==0, so no validity mask is needed.
  float mi = 0.0f;
  for (int m = 0; m < M; ++m) {
    const float* g = &gts[m * 4];
    mi = fmaxf(mi, iou_xywh(px, py, pw, ph, g[0], g[1], g[2], g[3]));
  }
  max_iou[((size_t)b * HWDIM + hw) * ADIM + a] = mi;
  if (mi > IGNORE_T) atomicOr(&has_pos[b], 1);
}

// Last-writer-wins scan == atomicMax over object index m.
__global__ void yolo_winner(const int* __restrict__ cellA, const int* __restrict__ aidxA,
                            const int* __restrict__ num_obj, int* __restrict__ winner,
                            int B, int M) {
  int i = blockIdx.x * blockDim.x + threadIdx.x;
  if (i >= B * M) return;
  int b = i / M, m = i - b * M;
  if (m < num_obj[b]) {
    int cell = cellA[i], ai = aidxA[i];
    atomicMax(&winner[((size_t)b * HWDIM + cell) * ADIM + ai], m);
  }
}

__global__ void yolo_loss(const float* __restrict__ outputs, const float* __restrict__ anchors,
                          const float* __restrict__ gt, const int* __restrict__ gcls,
                          const float* __restrict__ max_iou, const int* __restrict__ winner,
                          const int* __restrict__ num_obj, const int* __restrict__ has_pos,
                          float* __restrict__ partials, int M) {
  int b = blockIdx.y;
  int idx = blockIdx.x * blockDim.x + threadIdx.x;
  float contrib = 0.0f;

  if (idx < ADIM * HWDIM) {
    int a  = idx / HWDIM;
    int hw = idx - a * HWDIM;
    const float* base = outputs + ((size_t)b * (ADIM * CHS) + (size_t)a * CHS) * HWDIM + hw;
    float conf = sigf(base[4 * HWDIM]);
    size_t slot = ((size_t)b * HWDIM + hw) * ADIM + a;
    float mi = max_iou[slot];
    int win = winner[slot];

    if (win >= 0) {
      // objectness: (conf - max_iou_at_slot)^2
      float d = conf - mi;
      contrib += d * d;
      // box loss
      float aw = anchors[2 * a], ah = anchors[2 * a + 1];
      const float* g = gt + ((size_t)b * M + win) * 4;
      float bt0 = g[0] - (float)(hw % WDIM);
      float bt1 = g[1] - (float)(hw / WDIM);
      float bt2 = g[2] / aw;
      float bt3 = g[3] / ah;
      float pd0 = sigf(base[0]);
      float pd1 = sigf(base[HWDIM]);
      float pd2 = expf(base[2 * HWDIM]);
      float pd3 = expf(base[3 * HWDIM]);
      float bs = 2.0f - (pd2 * aw / (float)WDIM) * (pd3 * ah / (float)HDIM);
      float s = sqrtf(bs);  // keep sqrt so bs<0 -> NaN, matching reference
      float e0 = (pd0 - bt0) * s, e1 = (pd1 - bt1) * s;
      float e2 = (pd2 - bt2) * s, e3 = (pd3 - bt3) * s;
      contrib += e0 * e0 + e1 * e1 + e2 * e2 + e3 * e3;
      // class cross-entropy (log-softmax over 20 logits)
      int cls = gcls[b * M + win];
      float mx = -3.4028234663852886e38f;
      float lcls = 0.0f;
#pragma unroll
      for (int k = 0; k < NCLS; ++k) {
        float l = base[(5 + k) * HWDIM];
        mx = fmaxf(mx, l);
        if (k == cls) lcls = l;
      }
      float se = 0.0f;
#pragma unroll
      for (int k = 0; k < NCLS; ++k) se += expf(base[(5 + k) * HWDIM] - mx);
      contrib += logf(se) + mx - lcls;
    } else {
      // noobj mask: 1 unless (has_pos && mi >= thresh) or num_obj == 0
      if (num_obj[b] != 0 && !(has_pos[b] && mi >= IGNORE_T))
        contrib += conf * conf;
    }
  }

  // Block reduction (all 256 threads reach here; EXEC all-ones for WMMA).
  float wsum = wave_reduce_sum(contrib);
  __shared__ float smem[8];
  int lane = threadIdx.x & 31, wid = threadIdx.x >> 5;
  if (lane == 0) smem[wid] = wsum;
  __syncthreads();
  if (threadIdx.x == 0) {
    float s = 0.0f;
#pragma unroll
    for (int i = 0; i < 8; ++i) s += smem[i];
    partials[blockIdx.y * gridDim.x + blockIdx.x] = s;
  }
}

// Single block, fixed-order deterministic final sum.
__global__ void yolo_final(const float* __restrict__ partials, int n, float invB,
                           float* __restrict__ out) {
  float s = 0.0f;
  for (int i = threadIdx.x; i < n; i += blockDim.x) s += partials[i];
  float wsum = wave_reduce_sum(s);
  __shared__ float smem[8];
  int lane = threadIdx.x & 31, wid = threadIdx.x >> 5;
  if (lane == 0) smem[wid] = wsum;
  __syncthreads();
  if (threadIdx.x == 0) {
    float t = 0.0f;
#pragma unroll
    for (int i = 0; i < 8; ++i) t += smem[i];
    out[0] = t * invB;
  }
}

extern "C" void kernel_launch(void* const* d_in, const int* in_sizes, int n_in,
                              void* d_out, int out_size, void* d_ws, size_t ws_size,
                              hipStream_t stream) {
  const float* outputs = (const float*)d_in[0];
  const float* targets = (const float*)d_in[1];
  const float* anchors = (const float*)d_in[2];

  int B = in_sizes[0] / (ADIM * CHS * HWDIM);   // 128
  int M = in_sizes[1] / (B * 5);                // 64
  int ntot = B * HWDIM * ADIM;                  // 432640

  char* p = (char*)d_ws;
  size_t off = 0;
  auto take = [&](size_t bytes) -> void* {
    void* r = p + off;
    off += (bytes + 255) & ~(size_t)255;
    return r;
  };
  float* gt      = (float*)take((size_t)B * M * 4 * sizeof(float));
  int*   gcls    = (int*)  take((size_t)B * M * sizeof(int));
  int*   cellA   = (int*)  take((size_t)B * M * sizeof(int));
  int*   aidxA   = (int*)  take((size_t)B * M * sizeof(int));
  int*   numobj  = (int*)  take((size_t)B * sizeof(int));
  int*   haspos  = (int*)  take((size_t)B * sizeof(int));
  float* maxiou  = (float*)take((size_t)ntot * sizeof(float));
  int*   winner  = (int*)  take((size_t)ntot * sizeof(int));
  int gx = (ADIM * HWDIM + 255) / 256;          // 14
  float* partials = (float*)take((size_t)gx * B * sizeof(float));
  (void)ws_size; (void)n_in; (void)out_size;

  yolo_init<<<(ntot + 255) / 256, 256, 0, stream>>>(winner, haspos, ntot, B);
  yolo_prep<<<B, M, 0, stream>>>(targets, anchors, gt, gcls, cellA, aidxA, numobj, M);
  dim3 grid(gx, B);
  yolo_maxiou<<<grid, 256, 0, stream>>>(outputs, anchors, gt, maxiou, haspos, M);
  yolo_winner<<<(B * M + 255) / 256, 256, 0, stream>>>(cellA, aidxA, numobj, winner, B, M);
  yolo_loss<<<grid, 256, 0, stream>>>(outputs, anchors, gt, gcls, maxiou, winner,
                                      numobj, haspos, partials, M);
  yolo_final<<<1, 256, 0, stream>>>(partials, gx * B, 1.0f / (float)B, (float*)d_out);
}
